// _SpatialSelfAttention_3994319586122
// MI455X (gfx1250) — compile-verified, matrix operand order checked by reading, not run on hardware
//
#include <hip/hip_runtime.h>
#include <math.h>

typedef __attribute__((ext_vector_type(16))) __bf16 v16bf;
typedef __attribute__((ext_vector_type(8)))  float  v8f;

#define C_DIM 512
#define N_SP  4096
#define GCH   16
#define QKV_N 1536

// workspace layout (bytes)
#define OFF_WQKV ((size_t)0)
#define OFF_BQKV ((size_t)(3*512*512*2))
#define OFF_WP   (OFF_BQKV + (size_t)1536*4)
#define OFF_HF   (OFF_WP + (size_t)512*512*2)
#define OFF_QKV  (OFF_HF + (size_t)4*4096*512*2)
#define OFF_O    (OFF_QKV + (size_t)4*4096*1536*2)

// LDS byte address for async-DMA ops, derived from the real shared pointer so
// the LDS object escapes (ptrtoint = capture) and asm memory clobbers alias it.
__device__ __forceinline__ unsigned lds_u32(const void* p) {
  return (unsigned)(unsigned long long)p;
}

// ---------------------------------------------------------------------------
// 1) pack weights/biases
// ---------------------------------------------------------------------------
__global__ void prep_kernel(const float* __restrict__ wq, const float* __restrict__ wk,
                            const float* __restrict__ wv, const float* __restrict__ wp,
                            const float* __restrict__ bq, const float* __restrict__ bk,
                            const float* __restrict__ bv,
                            __bf16* __restrict__ wqkv, __bf16* __restrict__ wpb,
                            float* __restrict__ bqkv) {
  int t = blockIdx.x * blockDim.x + threadIdx.x;
  const int CC = C_DIM * C_DIM;
  if (t < 3 * CC) {
    int o = t / C_DIM, c = t - o * C_DIM;
    float v;
    if (o < C_DIM)          v = wq[o * C_DIM + c];
    else if (o < 2 * C_DIM) v = wk[(o - C_DIM) * C_DIM + c];
    else                    v = wv[(o - 2 * C_DIM) * C_DIM + c];
    wqkv[t] = (__bf16)v;
  }
  if (t < CC) wpb[t] = (__bf16)wp[t];
  if (t < QKV_N) {
    float v = (t < C_DIM) ? bq[t] : (t < 2 * C_DIM ? bk[t - C_DIM] : bv[t - 2 * C_DIM]);
    bqkv[t] = v;
  }
}

// ---------------------------------------------------------------------------
// 2) GroupNorm: x[b][c][n] f32 -> hf[b][n][c] bf16 (transposed GEMM-A layout)
// ---------------------------------------------------------------------------
__global__ __launch_bounds__(256) void groupnorm_kernel(const float* __restrict__ x,
                                                        const float* __restrict__ gamma,
                                                        const float* __restrict__ beta,
                                                        __bf16* __restrict__ hf) {
  __shared__ float rs[256], rss[256];
  __shared__ float s_mean, s_rstd;
  int b = blockIdx.x >> 5;
  int g = blockIdx.x & 31;
  int t = threadIdx.x;
  const float* xg = x + ((size_t)b * C_DIM + g * GCH) * N_SP;
  float s = 0.f, ss = 0.f;
  for (int i = t; i < GCH * N_SP; i += 256) {
    float v = xg[i];
    s += v; ss += v * v;
  }
  rs[t] = s; rss[t] = ss; __syncthreads();
  for (int off = 128; off; off >>= 1) {
    if (t < off) { rs[t] += rs[t + off]; rss[t] += rss[t + off]; }
    __syncthreads();
  }
  if (t == 0) {
    float inv = 1.0f / (float)(GCH * N_SP);
    float mean = rs[0] * inv;
    float var  = rss[0] * inv - mean * mean;
    s_mean = mean;
    s_rstd = rsqrtf(var + 1e-6f);
  }
  __syncthreads();
  float mean = s_mean, rstd = s_rstd;
  for (int i = t; i < GCH * N_SP; i += 256) {
    int ch = i & (GCH - 1);
    int n  = i >> 4;
    int c  = g * GCH + ch;
    float v = xg[(size_t)ch * N_SP + n];
    float y = (v - mean) * rstd * gamma[c] + beta[c];
    hf[((size_t)b * N_SP + n) * C_DIM + c] = (__bf16)y;
  }
}

// ---------------------------------------------------------------------------
// GEMM core: 128x64 tile, BK=32, 8 waves x (2x2) WMMA, async-DMA double buffer
// LDS per buffer: A tile 128x32 bf16 (8192B) then B tile 64x32 (4096B)
// ---------------------------------------------------------------------------
#define TILE_BYTES 12288

// ---------------------------------------------------------------------------
// 3) QKV GEMM: hf[b][n][512] x wqkv[1536][512]^T + bias -> qkv[b][n][1536]
// ---------------------------------------------------------------------------
__global__ __launch_bounds__(256) void gemm_qkv_kernel(const __bf16* __restrict__ hf,
                                                       const __bf16* __restrict__ W,
                                                       const float* __restrict__ bias,
                                                       __bf16* __restrict__ out) {
  __shared__ __attribute__((aligned(64))) char gsm[2 * TILE_BYTES];
  asm volatile("" :: "v"((char*)gsm) : "memory");  // escape LDS for DMA aliasing
  int t = threadIdx.x;
  int b = blockIdx.z;
  int m0 = blockIdx.x * 128;
  int n0 = blockIdx.y * 64;
  int wave = t >> 5, lane = t & 31;
  int wm = wave >> 1, wn = wave & 1;
  int lr = lane & 15, lh = lane >> 4;
  const __bf16* Ab = hf + ((size_t)b * N_SP + m0) * C_DIM;
  v8f acc00 = {}, acc01 = {}, acc10 = {}, acc11 = {};

  int arow = t >> 1, a16 = (t & 1) << 4;     // A: 32B per thread
  int bcol = t >> 2, b8  = (t & 3) << 3;     // B: 16B per thread
  auto issue_tile = [&](int k0, int buf) {
    char* tile = gsm + buf * TILE_BYTES;
    unsigned abase = lds_u32(tile + (arow * 32 + a16) * 2);
    const __bf16* agp = Ab + (size_t)arow * C_DIM + k0 + a16;
    asm volatile("global_load_async_to_lds_b128 %0, %1, off\n\t"
                 "global_load_async_to_lds_b128 %0, %1, off offset:16"
                 :: "v"(abase), "v"(agp) : "memory");
    unsigned bbase = lds_u32(tile + 8192 + (bcol * 32 + b8) * 2);
    const __bf16* bgp = W + (size_t)(n0 + bcol) * C_DIM + k0 + b8;
    asm volatile("global_load_async_to_lds_b128 %0, %1, off"
                 :: "v"(bbase), "v"(bgp) : "memory");
  };

  issue_tile(0, 0);
  int buf = 0;
  for (int k0 = 0; k0 < C_DIM; k0 += 32, buf ^= 1) {
    if (k0 + 32 < C_DIM) {
      issue_tile(k0 + 32, buf ^ 1);
      asm volatile("s_wait_asynccnt 0x3" ::: "memory");   // prev tile's 3 ops done
    } else {
      asm volatile("s_wait_asynccnt 0x0" ::: "memory");
    }
    __syncthreads();
    const __bf16* Asb = (const __bf16*)(gsm + buf * TILE_BYTES);
    const __bf16* Bsb = Asb + 4096;
    v16bf a0 = *(const v16bf*)(Asb + (wm * 32 + lr) * 32 + lh * 16);
    v16bf a1 = *(const v16bf*)(Asb + (wm * 32 + 16 + lr) * 32 + lh * 16);
    v16bf b0 = *(const v16bf*)(Bsb + (wn * 32 + lr) * 32 + lh * 16);
    v16bf b1 = *(const v16bf*)(Bsb + (wn * 32 + 16 + lr) * 32 + lh * 16);
    acc00 = __builtin_amdgcn_wmma_f32_16x16x32_bf16(false, a0, false, b0, (short)0, acc00, false, false);
    acc01 = __builtin_amdgcn_wmma_f32_16x16x32_bf16(false, a0, false, b1, (short)0, acc01, false, false);
    acc10 = __builtin_amdgcn_wmma_f32_16x16x32_bf16(false, a1, false, b0, (short)0, acc10, false, false);
    acc11 = __builtin_amdgcn_wmma_f32_16x16x32_bf16(false, a1, false, b1, (short)0, acc11, false, false);
    __syncthreads();
  }

  v8f* accs[2][2] = {{&acc00, &acc01}, {&acc10, &acc11}};
  for (int i = 0; i < 2; i++)
    for (int j = 0; j < 2; j++) {
      int col = n0 + wn * 32 + j * 16 + lr;
      float bsv = bias[col];
      v8f av = *accs[i][j];
      #pragma unroll
      for (int v = 0; v < 8; v++) {
        int row = m0 + wm * 32 + i * 16 + v + (lh << 3);
        out[((size_t)b * N_SP + row) * QKV_N + col] = (__bf16)(av[v] + bsv);
      }
    }
}

// ---------------------------------------------------------------------------
// 4) Flash attention: qkv[b][n][1536] -> o[b][n][512] bf16
//    64 query rows / block, 512 thr = 16 wave32, 64 keys per iteration.
// ---------------------------------------------------------------------------
#define VT_STRIDE 80

__global__ __launch_bounds__(512) void attn_kernel(const __bf16* __restrict__ qkv,
                                                   __bf16* __restrict__ obuf) {
  extern __shared__ __attribute__((aligned(64))) char smem[];
  __bf16* Qs = (__bf16*)smem;                 // 64*512
  __bf16* Ks = Qs + 64 * 512;                 // 64*512
  __bf16* Vt = Ks + 64 * 512;                 // 512*80 (transposed V)
  float*  Ss = (float*)(Vt + 512 * VT_STRIDE);
  __bf16* Ps = (__bf16*)(Ss + 64 * 64);
  float*  crow = (float*)(Ps + 64 * 64);
  float*  lrow = crow + 64;
  float*  mrow = lrow + 64;
  asm volatile("" :: "v"((char*)smem) : "memory");  // escape LDS for DMA aliasing

  int t = threadIdx.x;
  int b = blockIdx.y;
  int m0 = blockIdx.x * 64;
  int wave = t >> 5, lane = t & 31;
  int lr = lane & 15, lh = lane >> 4;
  int wr = wave >> 2;
  int wc = wave & 3;
  const __bf16* qb = qkv + (size_t)b * N_SP * QKV_N;

  // async-load Q tile straight into LDS (4 x 32B per thread)
  #pragma unroll
  for (int ii = 0; ii < 4; ii++) {
    int e = ii * 512 + t;
    int row = e >> 5, d = (e & 31) << 4;
    unsigned ldsa = lds_u32(Qs + row * 512 + d);
    const __bf16* gp = qb + (size_t)(m0 + row) * QKV_N + d;
    asm volatile("global_load_async_to_lds_b128 %0, %1, off\n\t"
                 "global_load_async_to_lds_b128 %0, %1, off offset:16"
                 :: "v"(ldsa), "v"(gp) : "memory");
  }
  if (t < 64) { mrow[t] = -1e30f; lrow[t] = 0.f; }

  v8f acc[8] = {};
  const float scl = 0.044194173824159216f;  // 512^-0.5

  for (int j0 = 0; j0 < N_SP; j0 += 64) {
    // K tile: async DMA to LDS; V tile: load + transpose into Vt[d][j]
    #pragma unroll
    for (int ii = 0; ii < 4; ii++) {
      int e = ii * 512 + t;
      int row = e >> 5, d = (e & 31) << 4;
      const __bf16* src = qb + (size_t)(j0 + row) * QKV_N;
      unsigned ldsa = lds_u32(Ks + row * 512 + d);
      asm volatile("global_load_async_to_lds_b128 %0, %1, off\n\t"
                   "global_load_async_to_lds_b128 %0, %1, off offset:16"
                   :: "v"(ldsa), "v"(src + 512 + d) : "memory");
      v16bf vv = *(const v16bf*)(src + 1024 + d);
      #pragma unroll
      for (int q = 0; q < 16; q++) Vt[(d + q) * VT_STRIDE + row] = vv[q];
      if (j0 + 64 < N_SP)
        __builtin_prefetch(src + (size_t)64 * QKV_N + 1024 + d, 0, 1);
    }
    asm volatile("s_wait_asynccnt 0x0" ::: "memory");
    __syncthreads();

    // S = scale * Q K^T : each wave owns one 16x16 tile of the 64x64 block
    {
      v8f sa = {};
      int qrow = wr * 16 + lr;
      int krow = wc * 16 + lr;
      #pragma unroll
      for (int kk = 0; kk < C_DIM; kk += 32) {
        v16bf a  = *(const v16bf*)&Qs[qrow * 512 + kk + lh * 16];
        v16bf bb = *(const v16bf*)&Ks[krow * 512 + kk + lh * 16];
        sa = __builtin_amdgcn_wmma_f32_16x16x32_bf16(false, a, false, bb, (short)0, sa, false, false);
      }
      #pragma unroll
      for (int v = 0; v < 8; v++) {
        int sr = wr * 16 + v + (lh << 3);
        int sc = wc * 16 + lr;
        Ss[sr * 64 + sc] = sa[v] * scl;
      }
    }
    __syncthreads();

    // online softmax: 8 threads per row, shfl_xor subgroup reductions
    {
      int r = t >> 3;
      int sub = t & 7;
      const float* srow = &Ss[r * 64 + sub * 8];
      float4 f0 = *(const float4*)(srow);
      float4 f1 = *(const float4*)(srow + 4);
      float mx = fmaxf(fmaxf(fmaxf(f0.x, f0.y), fmaxf(f0.z, f0.w)),
                       fmaxf(fmaxf(f1.x, f1.y), fmaxf(f1.z, f1.w)));
      mx = fmaxf(mx, __shfl_xor(mx, 1, 32));
      mx = fmaxf(mx, __shfl_xor(mx, 2, 32));
      mx = fmaxf(mx, __shfl_xor(mx, 4, 32));
      float mold = mrow[r];
      float mnew = fmaxf(mold, mx);
      float sum = 0.f;
      __bf16* prow = &Ps[r * 64 + sub * 8];
      #pragma unroll
      for (int c2 = 0; c2 < 8; c2++) {
        float p = __expf(srow[c2] - mnew);
        prow[c2] = (__bf16)p;
        sum += p;
      }
      sum += __shfl_xor(sum, 1, 32);
      sum += __shfl_xor(sum, 2, 32);
      sum += __shfl_xor(sum, 4, 32);
      if (sub == 0) {
        float ce = __expf(mold - mnew);
        crow[r] = ce;
        lrow[r] = lrow[r] * ce + sum;
        mrow[r] = mnew;
      }
    }
    __syncthreads();

    // rescale accumulators, then O += P * V
    #pragma unroll
    for (int v = 0; v < 8; v++) {
      float ce = crow[wr * 16 + v + (lh << 3)];
      #pragma unroll
      for (int tt = 0; tt < 8; tt++) acc[tt][v] *= ce;
    }
    #pragma unroll
    for (int kh = 0; kh < 2; kh++) {
      int kk = kh * 32;
      v16bf a = *(const v16bf*)&Ps[(wr * 16 + lr) * 64 + kk + lh * 16];
      #pragma unroll
      for (int tt = 0; tt < 8; tt++) {
        int dcol = wc * 128 + tt * 16 + lr;
        v16bf bb = *(const v16bf*)&Vt[dcol * VT_STRIDE + kk + lh * 16];
        acc[tt] = __builtin_amdgcn_wmma_f32_16x16x32_bf16(false, a, false, bb, (short)0, acc[tt], false, false);
      }
    }
    __syncthreads();
  }

  #pragma unroll
  for (int v = 0; v < 8; v++) {
    int row = wr * 16 + v + (lh << 3);
    float inv = 1.0f / lrow[row];
    #pragma unroll
    for (int tt = 0; tt < 8; tt++) {
      int dcol = wc * 128 + tt * 16 + lr;
      obuf[((size_t)b * N_SP + m0 + row) * C_DIM + dcol] = (__bf16)(acc[tt][v] * inv);
    }
  }
}

// ---------------------------------------------------------------------------
// 5) Proj GEMM + residual: o[b][n][512] x wp^T + bp -> transpose + x -> out f32
// ---------------------------------------------------------------------------
__global__ __launch_bounds__(256) void gemm_proj_kernel(const __bf16* __restrict__ ob,
                                                        const __bf16* __restrict__ W,
                                                        const float* __restrict__ bias,
                                                        const float* __restrict__ x,
                                                        float* __restrict__ out) {
  __shared__ __attribute__((aligned(64))) char gsm[32768];  // 2 tile buffers; reused as f32 C stage
  asm volatile("" :: "v"((char*)gsm) : "memory");  // escape LDS for DMA aliasing
  int t = threadIdx.x;
  int b = blockIdx.z;
  int m0 = blockIdx.x * 128;
  int n0 = blockIdx.y * 64;
  int wave = t >> 5, lane = t & 31;
  int wm = wave >> 1, wn = wave & 1;
  int lr = lane & 15, lh = lane >> 4;
  const __bf16* Ab = ob + ((size_t)b * N_SP + m0) * C_DIM;
  v8f acc00 = {}, acc01 = {}, acc10 = {}, acc11 = {};

  int arow = t >> 1, a16 = (t & 1) << 4;
  int bcol = t >> 2, b8  = (t & 3) << 3;
  auto issue_tile = [&](int k0, int buf) {
    char* tile = gsm + buf * TILE_BYTES;
    unsigned abase = lds_u32(tile + (arow * 32 + a16) * 2);
    const __bf16* agp = Ab + (size_t)arow * C_DIM + k0 + a16;
    asm volatile("global_load_async_to_lds_b128 %0, %1, off\n\t"
                 "global_load_async_to_lds_b128 %0, %1, off offset:16"
                 :: "v"(abase), "v"(agp) : "memory");
    unsigned bbase = lds_u32(tile + 8192 + (bcol * 32 + b8) * 2);
    const __bf16* bgp = W + (size_t)(n0 + bcol) * C_DIM + k0 + b8;
    asm volatile("global_load_async_to_lds_b128 %0, %1, off"
                 :: "v"(bbase), "v"(bgp) : "memory");
  };

  issue_tile(0, 0);
  int buf = 0;
  for (int k0 = 0; k0 < C_DIM; k0 += 32, buf ^= 1) {
    if (k0 + 32 < C_DIM) {
      issue_tile(k0 + 32, buf ^ 1);
      asm volatile("s_wait_asynccnt 0x3" ::: "memory");
    } else {
      asm volatile("s_wait_asynccnt 0x0" ::: "memory");
    }
    __syncthreads();
    const __bf16* Asb = (const __bf16*)(gsm + buf * TILE_BYTES);
    const __bf16* Bsb = Asb + 4096;
    v16bf a0 = *(const v16bf*)(Asb + (wm * 32 + lr) * 32 + lh * 16);
    v16bf a1 = *(const v16bf*)(Asb + (wm * 32 + 16 + lr) * 32 + lh * 16);
    v16bf b0 = *(const v16bf*)(Bsb + (wn * 32 + lr) * 32 + lh * 16);
    v16bf b1 = *(const v16bf*)(Bsb + (wn * 32 + 16 + lr) * 32 + lh * 16);
    acc00 = __builtin_amdgcn_wmma_f32_16x16x32_bf16(false, a0, false, b0, (short)0, acc00, false, false);
    acc01 = __builtin_amdgcn_wmma_f32_16x16x32_bf16(false, a0, false, b1, (short)0, acc01, false, false);
    acc10 = __builtin_amdgcn_wmma_f32_16x16x32_bf16(false, a1, false, b0, (short)0, acc10, false, false);
    acc11 = __builtin_amdgcn_wmma_f32_16x16x32_bf16(false, a1, false, b1, (short)0, acc11, false, false);
    __syncthreads();
  }

  // stage C tile in LDS (f32), then transposed coalesced write-out + residual
  float* Cs = (float*)gsm;   // [128][64]
  v8f* accs[2][2] = {{&acc00, &acc01}, {&acc10, &acc11}};
  for (int i = 0; i < 2; i++)
    for (int j = 0; j < 2; j++) {
      int cl = wn * 32 + j * 16 + lr;
      float bsv = bias[n0 + cl];
      v8f av = *accs[i][j];
      #pragma unroll
      for (int v = 0; v < 8; v++) {
        int rl = wm * 32 + i * 16 + v + (lh << 3);
        Cs[rl * 64 + cl] = av[v] + bsv;
      }
    }
  __syncthreads();
  for (int it = 0; it < 32; it++) {
    int cl = (t >> 7) + it * 2;
    int rl = t & 127;
    size_t idx = ((size_t)b * C_DIM + n0 + cl) * N_SP + m0 + rl;
    out[idx] = x[idx] + Cs[rl * 64 + cl];
  }
}

// ---------------------------------------------------------------------------
extern "C" void kernel_launch(void* const* d_in, const int* in_sizes, int n_in,
                              void* d_out, int out_size, void* d_ws, size_t ws_size,
                              hipStream_t stream) {
  const float* x     = (const float*)d_in[0];
  const float* gamma = (const float*)d_in[1];
  const float* beta  = (const float*)d_in[2];
  const float* wq    = (const float*)d_in[3];
  const float* bq    = (const float*)d_in[4];
  const float* wk    = (const float*)d_in[5];
  const float* bk    = (const float*)d_in[6];
  const float* wv    = (const float*)d_in[7];
  const float* bv    = (const float*)d_in[8];
  const float* wp    = (const float*)d_in[9];
  const float* bp    = (const float*)d_in[10];
  float* out = (float*)d_out;

  char* ws = (char*)d_ws;
  __bf16* wqkv = (__bf16*)(ws + OFF_WQKV);
  float*  bqkv = (float*)(ws + OFF_BQKV);
  __bf16* wpb  = (__bf16*)(ws + OFF_WP);
  __bf16* hf   = (__bf16*)(ws + OFF_HF);
  __bf16* qkv  = (__bf16*)(ws + OFF_QKV);
  __bf16* obuf = (__bf16*)(ws + OFF_O);

  prep_kernel<<<3072, 256, 0, stream>>>(wq, wk, wv, wp, bq, bk, bv, wqkv, wpb, bqkv);
  groupnorm_kernel<<<128, 256, 0, stream>>>(x, gamma, beta, hf);
  gemm_qkv_kernel<<<dim3(32, 24, 4), 256, 0, stream>>>(hf, wqkv, bqkv, qkv);

  size_t attn_smem = (size_t)64 * 512 * 2 * 2      // Qs + Ks
                   + (size_t)512 * VT_STRIDE * 2   // Vt
                   + (size_t)64 * 64 * 4           // Ss
                   + (size_t)64 * 64 * 2           // Ps
                   + 3 * 64 * 4;                   // crow + lrow + mrow
  attn_kernel<<<dim3(64, 4), 512, attn_smem, stream>>>(qkv, obuf);

  gemm_proj_kernel<<<dim3(32, 8, 4), 256, 0, stream>>>(obuf, wpb, bp, x, out);
}
